// VisionTransformerForMaskedImageModeling_59725815218674
// MI455X (gfx1250) — matrix operand order repaired and checked by reading, not compiled, not run on hardware
//
#include <hip/hip_runtime.h>
#include <hip/hip_bf16.h>

// ---------------------------------------------------------------------------
// ViT visual embed for MI455X (gfx1250, wave32, WMMA).
// GEMM: one wave owns a 32x64 output tile (2 M-tiles x 4 N-tiles), 8
// v_wmma_f32_16x16x32_bf16 per K-step; fragments pre-swizzled in workspace so
// every lane issues contiguous global_load_b128 from L2-resident buffers.
// ---------------------------------------------------------------------------

typedef __attribute__((ext_vector_type(16))) __bf16 v16bf;
typedef __attribute__((ext_vector_type(8)))  float  v8f;

#define B_   128
#define C_   768
#define P_   14
#define NP_  196      // P*P
#define L_   145
#define LP1_ 146
#define MT_  10       // ceil(145/16) row tiles (rows 145..159 zero-padded)
#define KS_  24       // 768 / 32
#define NT_  48       // 768 / 16
#define MG_  5        // M-tile groups of 2
#define NG_  12       // N-tile groups of 4

// ------------------------- threefry-2x32 (JAX exact) ------------------------
__device__ __forceinline__ unsigned rotl32(unsigned x, int r) {
  return (x << r) | (x >> (32 - r));
}
__device__ __forceinline__ void threefry2x32(unsigned k0, unsigned k1,
                                             unsigned& x0, unsigned& x1) {
  unsigned ks2 = k0 ^ k1 ^ 0x1BD11BDAu;
  x0 += k0; x1 += k1;
#define TF_G(r0, r1, r2, r3, a0, a1, inc)                                      \
  x0 += x1; x1 = rotl32(x1, r0); x1 ^= x0;                                     \
  x0 += x1; x1 = rotl32(x1, r1); x1 ^= x0;                                     \
  x0 += x1; x1 = rotl32(x1, r2); x1 ^= x0;                                     \
  x0 += x1; x1 = rotl32(x1, r3); x1 ^= x0;                                     \
  x0 += (a0); x1 += (a1) + (inc);
  TF_G(13, 15, 26, 6, k1, ks2, 1u)
  TF_G(17, 29, 16, 24, ks2, k0, 2u)
  TF_G(13, 15, 26, 6, k0, k1, 3u)
  TF_G(17, 29, 16, 24, k1, ks2, 4u)
  TF_G(13, 15, 26, 6, ks2, k0, 5u)
#undef TF_G
}
// noise[b,p] == jax.random.uniform(key(42), (128,196))[b,p]
__device__ __forceinline__ float jax_noise(int n) {
  const int half = (B_ * NP_) / 2;  // 12544
  bool second = n >= half;
  unsigned j = second ? (unsigned)(n - half) : (unsigned)n;
  unsigned x0 = j, x1 = j + (unsigned)half;
  threefry2x32(0u, 42u, x0, x1);
  unsigned bits = second ? x1 : x0;
  return __uint_as_float((bits >> 9) | 0x3f800000u) - 1.0f;
}

// ---------------- 1. valid rows/cols per sample (stride-16 probe) -----------
__global__ void hw_kernel(const float* __restrict__ img, int* __restrict__ hw) {
  int b = blockIdx.x * blockDim.x + threadIdx.x;
  if (b >= B_) return;
  int h = 0, w = 0;
  for (int i = 0; i < P_; ++i) {
    float sr = 0.f, sc = 0.f;
    for (int c = 0; c < 3; ++c) {
      size_t base = ((size_t)(b * 3 + c)) * 224 * 224;
      sr += img[base + (size_t)(i * 16) * 224];  // pixel (16i, 0)
      sc += img[base + (size_t)(i * 16)];        // pixel (0, 16i)
    }
    h += (sr != 0.f);
    w += (sc != 0.f);
  }
  hw[2 * b + 0] = h;
  hw[2 * b + 1] = w;
}

// ---------------- 2. ragged top-145 selection (stable desc sort) ------------
__global__ __launch_bounds__(256) void select_kernel(
    const int* __restrict__ hw, int* __restrict__ sel,
    float* __restrict__ mask_out, float* __restrict__ pidx_out) {
  int b = blockIdx.x;
  int t = threadIdx.x;
  __shared__ float key[NP_];
  int h = hw[2 * b], w = hw[2 * b + 1];
  if (t < NP_) {
    int py = t / P_, px = t % P_;
    float m = (py < h && px < w) ? 2.0f : 0.0f;
    key[t] = m + jax_noise(b * NP_ + t);
  }
  __syncthreads();
  if (t < NP_) {
    float k = key[t];
    int rank = 0;
    for (int q = 0; q < NP_; ++q) {
      float kq = key[q];
      rank += (kq > k) || (kq == k && q < t);
    }
    if (rank < L_) {
      sel[b * L_ + rank] = t;
      mask_out[b * LP1_ + 1 + rank] = (k >= 2.0f) ? 1.0f : 0.0f;
      pidx_out[(b * L_ + rank) * 2 + 0] = (float)(t / P_);
      pidx_out[(b * L_ + rank) * 2 + 1] = (float)(t % P_);
    }
  }
  if (t == 0) mask_out[b * LP1_] = 1.0f;  // cls always valid
}

// ---------------- 3a. weights -> bf16 B-fragment layout ---------------------
// B frag: lane&15 -> N; element i -> K = 16*(i>>3) + 8*(lane>>4) + (i&7)
__global__ void pack_w_kernel(const float* __restrict__ convw,
                              __bf16* __restrict__ Wws) {
  int t = blockIdx.x * blockDim.x + threadIdx.x;  // KS_*NT_*512 = 589824
  int i = t & 15;
  int lane = (t >> 4) & 31;
  int ntks = t >> 9;
  int nt = ntks % NT_;
  int ks = ntks / NT_;
  int K = ks * 32 + ((i >> 3) << 4) + ((lane >> 4) << 3) + (i & 7);
  int co = nt * 16 + (lane & 15);
  Wws[t] = (__bf16)convw[(size_t)co * C_ + K];  // B[k][n] = W[co=n][k]
}

// ---------------- 3b. selected patches -> bf16 A-fragment layout ------------
// A frag (16x32 bf16): lane&15 = row M; K = 16*(i>>3) + 8*(lane>>4)
//                                         + 2*((i>>1)&3) + (i&1)
// One thread handles element pair (2j, 2j+1): always two adjacent columns of
// the same image row -> one aligned float2 load, one packed b32 store.
__global__ void pack_a_kernel(const float* __restrict__ img,
                              const int* __restrict__ sel,
                              unsigned* __restrict__ Aws32) {
  int t = blockIdx.x * blockDim.x + threadIdx.x;  // B_*MT_*KS_*256 pairs
  int j = t & 7;            // pair index: elements 2j, 2j+1
  int lane = (t >> 3) & 31;
  int rest = t >> 8;
  int ks = rest % KS_;
  int bmt = rest / KS_;
  int mt = bmt % MT_;
  int b = bmt / MT_;
  int l = mt * 16 + (lane & 15);
  unsigned pk = 0u;
  if (l < L_) {
    int p = sel[b * L_ + l];
    int py = p / P_, px = p % P_;
    int K0 = ks * 32 + ((j >> 2) << 4) + ((lane >> 4) << 3) + ((j & 3) << 1);
    int ci = K0 >> 8;
    int ky = (K0 >> 4) & 15;
    int kx = K0 & 15;  // even -> float2-aligned
    size_t idx = (((size_t)(b * 3 + ci)) * 224 + (size_t)(py * 16 + ky)) * 224 +
                 (size_t)(px * 16 + kx);
    float2 f = *reinterpret_cast<const float2*>(&img[idx]);
    union { unsigned u; __bf16 h[2]; } u;
    u.h[0] = (__bf16)f.x;
    u.h[1] = (__bf16)f.y;
    pk = u.u;
  }
  Aws32[t] = pk;
}

// ---------------- 4. WMMA GEMM: tokens = A(160x768) * W^T + bias ------------
// One wave: 2 M-tiles x 4 N-tiles (32x64 output), 8 WMMAs per K-step.
__global__ __launch_bounds__(256) void gemm_kernel(
    const __bf16* __restrict__ Aws, const __bf16* __restrict__ Wws,
    const float* __restrict__ convb, float* __restrict__ out) {
  int lane = threadIdx.x & 31;
  int wv = __builtin_amdgcn_readfirstlane(threadIdx.x >> 5);  // uniform
  int gw = blockIdx.x * 8 + wv;              // global wave id
  int ng = gw % NG_;
  int mg = (gw / NG_) % MG_;
  int b = gw / (NG_ * MG_);

  const __bf16* A0 = Aws + ((size_t)(b * MT_ + 2 * mg) * KS_) * 512 + lane * 16;
  const __bf16* A1 = A0 + (size_t)KS_ * 512;
  const __bf16* Bb = Wws + (size_t)(4 * ng) * 512 + lane * 16;

  v8f acc[2][4] = {};
#pragma unroll 2
  for (int ks = 0; ks < KS_; ++ks) {
    v16bf a0 = *reinterpret_cast<const v16bf*>(A0 + (size_t)ks * 512);
    v16bf a1 = *reinterpret_cast<const v16bf*>(A1 + (size_t)ks * 512);
    const __bf16* bp = Bb + (size_t)ks * NT_ * 512;
    v16bf b0 = *reinterpret_cast<const v16bf*>(bp);
    v16bf b1 = *reinterpret_cast<const v16bf*>(bp + 512);
    v16bf b2 = *reinterpret_cast<const v16bf*>(bp + 1024);
    v16bf b3 = *reinterpret_cast<const v16bf*>(bp + 1536);
    acc[0][0] = __builtin_amdgcn_wmma_f32_16x16x32_bf16(false, a0, false, b0,
                                                        (short)0, acc[0][0],
                                                        false, false);
    acc[0][1] = __builtin_amdgcn_wmma_f32_16x16x32_bf16(false, a0, false, b1,
                                                        (short)0, acc[0][1],
                                                        false, false);
    acc[0][2] = __builtin_amdgcn_wmma_f32_16x16x32_bf16(false, a0, false, b2,
                                                        (short)0, acc[0][2],
                                                        false, false);
    acc[0][3] = __builtin_amdgcn_wmma_f32_16x16x32_bf16(false, a0, false, b3,
                                                        (short)0, acc[0][3],
                                                        false, false);
    acc[1][0] = __builtin_amdgcn_wmma_f32_16x16x32_bf16(false, a1, false, b0,
                                                        (short)0, acc[1][0],
                                                        false, false);
    acc[1][1] = __builtin_amdgcn_wmma_f32_16x16x32_bf16(false, a1, false, b1,
                                                        (short)0, acc[1][1],
                                                        false, false);
    acc[1][2] = __builtin_amdgcn_wmma_f32_16x16x32_bf16(false, a1, false, b2,
                                                        (short)0, acc[1][2],
                                                        false, false);
    acc[1][3] = __builtin_amdgcn_wmma_f32_16x16x32_bf16(false, a1, false, b3,
                                                        (short)0, acc[1][3],
                                                        false, false);
  }

  int lane15 = lane & 15;
  int mhi = (lane >> 4) * 8;  // VGPR r: lanes 0-15 -> M=r, lanes 16-31 -> M=r+8
  float bias[4];
#pragma unroll
  for (int n = 0; n < 4; ++n) bias[n] = convb[(4 * ng + n) * 16 + lane15];

  if (mg < MG_ - 1) {  // scalar branch: all 32 rows valid
#pragma unroll
    for (int m = 0; m < 2; ++m) {
      size_t rowbase = ((size_t)b * LP1_ + 1 + (2 * mg + m) * 16 + mhi) * C_;
#pragma unroll
      for (int n = 0; n < 4; ++n) {
        int co = (4 * ng + n) * 16 + lane15;
#pragma unroll
        for (int r = 0; r < 8; ++r)
          out[rowbase + (size_t)r * C_ + co] = acc[m][n][r] + bias[n];
      }
    }
  } else {  // mg==4: mt=8 fully valid, mt=9 only row 144 (r==0, lanes 0-15)
#pragma unroll
    for (int m = 0; m < 2; ++m) {
#pragma unroll
      for (int n = 0; n < 4; ++n) {
        int co = (4 * ng + n) * 16 + lane15;
#pragma unroll
        for (int r = 0; r < 8; ++r) {
          int l = (2 * mg + m) * 16 + r + mhi;
          if (l < L_)
            out[((size_t)b * LP1_ + 1 + l) * C_ + co] = acc[m][n][r] + bias[n];
        }
      }
    }
  }
}

// ---------------- 5. pos embed (bilinear, align_corners) + cls row ----------
__global__ __launch_bounds__(256) void pos_add_kernel(
    const float* __restrict__ pe, const float* __restrict__ cls,
    const int* __restrict__ hw, const int* __restrict__ sel,
    float* __restrict__ out) {
  int blk = blockIdx.x;
  int t = threadIdx.x;
  int b = blk / LP1_;
  int l1 = blk % LP1_;
  float* row = out + ((size_t)b * LP1_ + l1) * C_;
  if (l1 == 0) {
    for (int c = t; c < C_; c += 256) row[c] = cls[c] + pe[c];
    return;
  }
  int l = l1 - 1;
  int p = sel[b * L_ + l];
  int i = p / P_, j = p % P_;
  int h = hw[2 * b], w = hw[2 * b + 1];
  if (i >= h || j >= w) return;  // zero-padded pos region
  float hf = (float)max(h - 1, 1), wf = (float)max(w - 1, 1);
  float sy = fminf(fmaxf((float)i * 13.0f / hf, 0.f), 13.f);
  float sx = fminf(fmaxf((float)j * 13.0f / wf, 0.f), 13.f);
  int y0 = (int)floorf(sy), x0 = (int)floorf(sx);
  int y1 = min(y0 + 1, 13), x1 = min(x0 + 1, 13);
  float fy = sy - (float)y0, fx = sx - (float)x0;
  const float* p00 = pe + (size_t)(1 + y0 * P_ + x0) * C_;
  const float* p01 = pe + (size_t)(1 + y0 * P_ + x1) * C_;
  const float* p10 = pe + (size_t)(1 + y1 * P_ + x0) * C_;
  const float* p11 = pe + (size_t)(1 + y1 * P_ + x1) * C_;
  for (int c = t; c < C_; c += 256) {
    float top = p00[c] * (1.f - fx) + p01[c] * fx;
    float bot = p10[c] * (1.f - fx) + p11[c] * fx;
    row[c] += top * (1.f - fy) + bot * fy;
  }
}

// ---------------- 6. LayerNorm (in place, one block per token) --------------
__global__ __launch_bounds__(256) void ln_kernel(float* __restrict__ out,
                                                 const float* __restrict__ sc,
                                                 const float* __restrict__ bi) {
  int t = threadIdx.x;
  float* row = out + (size_t)blockIdx.x * C_;
  __shared__ float red[256];
  float v0 = row[t], v1 = row[t + 256], v2 = row[t + 512];
  red[t] = v0 + v1 + v2;
  __syncthreads();
  for (int o = 128; o > 0; o >>= 1) {
    if (t < o) red[t] += red[t + o];
    __syncthreads();
  }
  float mu = red[0] * (1.0f / C_);
  __syncthreads();
  float d0 = v0 - mu, d1 = v1 - mu, d2 = v2 - mu;
  red[t] = d0 * d0 + d1 * d1 + d2 * d2;
  __syncthreads();
  for (int o = 128; o > 0; o >>= 1) {
    if (t < o) red[t] += red[t + o];
    __syncthreads();
  }
  float inv = rsqrtf(red[0] * (1.0f / C_) + 1e-6f);
  row[t]       = d0 * inv * sc[t]       + bi[t];
  row[t + 256] = d1 * inv * sc[t + 256] + bi[t + 256];
  row[t + 512] = d2 * inv * sc[t + 512] + bi[t + 512];
}

// ---------------------------------------------------------------------------
extern "C" void kernel_launch(void* const* d_in, const int* in_sizes, int n_in,
                              void* d_out, int out_size, void* d_ws,
                              size_t ws_size, hipStream_t stream) {
  const float* images = (const float*)d_in[0];  // [128,3,224,224]
  const float* conv_w = (const float*)d_in[1];  // [768,3,16,16]
  const float* conv_b = (const float*)d_in[2];  // [768]
  const float* pos_e  = (const float*)d_in[3];  // [1,197,768]
  const float* cls_t  = (const float*)d_in[4];  // [1,1,768]
  const float* ln_s   = (const float*)d_in[5];  // [768]
  const float* ln_b   = (const float*)d_in[6];  // [768]

  // d_out: tokens [B,146,768] ++ mask [B,146] ++ pidx [B,145,2] (as float)
  float* out_tok  = (float*)d_out;
  float* out_mask = out_tok + (size_t)B_ * LP1_ * C_;
  float* out_pidx = out_mask + (size_t)B_ * LP1_;

  // workspace: hw ints, sel ints, bf16 W fragments, bf16 A fragments
  char* ws = (char*)d_ws;
  int* hw = (int*)ws;                    // 2*B ints
  int* sel = hw + 2 * B_;                // B*L ints
  size_t off = ((size_t)(2 * B_ + B_ * L_) * sizeof(int) + 255) & ~(size_t)255;
  __bf16* Wws = (__bf16*)(ws + off);     // 768*768 bf16
  off += (size_t)C_ * C_ * sizeof(__bf16);
  off = (off + 255) & ~(size_t)255;
  __bf16* Aws = (__bf16*)(ws + off);     // B*MT*KS*512 bf16 (~31.5 MB)

  hw_kernel<<<1, 128, 0, stream>>>(images, hw);
  select_kernel<<<B_, 256, 0, stream>>>(hw, sel, out_mask, out_pidx);
  pack_w_kernel<<<(KS_ * NT_ * 512) / 256, 256, 0, stream>>>(conv_w, Wws);
  pack_a_kernel<<<(B_ * MT_ * KS_ * 256) / 256, 256, 0, stream>>>(
      images, sel, (unsigned*)Aws);
  gemm_kernel<<<(B_ * MG_ * NG_) / 8, 256, 0, stream>>>(Aws, Wws, conv_b,
                                                        out_tok);
  pos_add_kernel<<<B_ * LP1_, 256, 0, stream>>>(pos_e, cls_t, hw, sel, out_tok);
  ln_kernel<<<B_ * LP1_, 256, 0, stream>>>(out_tok, ln_s, ln_b);
}
